// DecoderLayer_14628658610272
// MI455X (gfx1250) — matrix operand order repaired and checked by reading, not compiled, not run on hardware
//
#include <hip/hip_runtime.h>
#include <hip/hip_bf16.h>
#include <math.h>

// ---------------------------------------------------------------------------
// CDNA5 (gfx1250) decoder layer:
//  - all matmul paths use v_wmma_f32_16x16x32_bf16 (f32 accumulate)
//  - weights/activations pre-converted to bf16 once; GEMMs consume bf16 and
//    produce bf16 when the consumer is another WMMA kernel
//  - LDS staging via global_load_async_to_lds_b128 (ASYNCcnt), DOUBLE-BUFFERED
//    so the async engine fills tile t+1 while WMMA consumes tile t
//  - softmax row reductions use DPP16 ROW_XMASK butterflies (VALU only)
// ---------------------------------------------------------------------------

typedef __attribute__((ext_vector_type(16))) __bf16    v16bf;
typedef __attribute__((ext_vector_type(8)))  float     v8f;
typedef __attribute__((ext_vector_type(4)))  float     f32x4;
typedef __attribute__((ext_vector_type(4)))  unsigned  u32x4;
typedef __attribute__((ext_vector_type(4)))  int       i32x4;

union V16U {
    v16bf          v;
    unsigned short s[16];
    u32x4          q[2];
};
union U8x16 {
    u32x4          q;
    unsigned short s[8];
};

__device__ __forceinline__ unsigned short f2bf(float f) {
    unsigned int u = __float_as_uint(f);
    u += 0x7FFFu + ((u >> 16) & 1u);
    return (unsigned short)(u >> 16);
}

// pack two f32 -> two bf16 (RNE) in one dword via v_perm_b32
__device__ __forceinline__ unsigned rne2(float lo, float hi) {
    unsigned ul = __float_as_uint(lo), uh = __float_as_uint(hi);
    ul += 0x7FFFu + ((ul >> 16) & 1u);
    uh += 0x7FFFu + ((uh >> 16) & 1u);
    return __builtin_amdgcn_perm(uh, ul, 0x07060302u);
}

__device__ __forceinline__ v8f wmma_bf16(v16bf a, v16bf b, v8f c) {
    return __builtin_amdgcn_wmma_f32_16x16x32_bf16(
        false, a, false, b, (short)0, c, false, false);
}

// ---- DPP16 ROW_XMASK butterfly reductions over 16-lane rows (VALU only) ----
#define DPP_XOR(x, n) \
    __int_as_float(__builtin_amdgcn_update_dpp(0, __float_as_int(x), 0x160 | (n), 0xf, 0xf, true))

__device__ __forceinline__ float row16_max(float x) {
    x = fmaxf(x, DPP_XOR(x, 1));
    x = fmaxf(x, DPP_XOR(x, 2));
    x = fmaxf(x, DPP_XOR(x, 4));
    x = fmaxf(x, DPP_XOR(x, 8));
    return x;
}
__device__ __forceinline__ float row16_sum(float x) {
    x += DPP_XOR(x, 1);
    x += DPP_XOR(x, 2);
    x += DPP_XOR(x, 4);
    x += DPP_XOR(x, 8);
    return x;
}

// ---- async 16B global->LDS copy (gfx1250 GLOBAL_LOAD_ASYNC_TO_LDS_B128) ----
#if __has_builtin(__builtin_amdgcn_global_load_async_to_lds_b128)
#define HAVE_ASYNC_LDS 1
#else
#define HAVE_ASYNC_LDS 0
#endif

#if HAVE_ASYNC_LDS
typedef __attribute__((address_space(1))) i32x4 as1_i32x4;
typedef __attribute__((address_space(3))) i32x4 as3_i32x4;
#endif

__device__ __forceinline__ void cp16(unsigned short* ldst, const unsigned short* gsrc) {
#if HAVE_ASYNC_LDS
    // generic->AS1 keeps the numeric address; generic LDS address truncates to
    // the LDS offset in the low 32 bits (ISA 10.2 aperture rules)
    __builtin_amdgcn_global_load_async_to_lds_b128(
        (as1_i32x4*)(size_t)gsrc,
        (as3_i32x4*)(unsigned)(size_t)ldst,
        0, 0);
#else
    *(u32x4*)ldst = *(const u32x4*)gsrc;
#endif
}

__device__ __forceinline__ void async_join() {
#if HAVE_ASYNC_LDS
#if __has_builtin(__builtin_amdgcn_s_wait_asynccnt)
    __builtin_amdgcn_s_wait_asynccnt(0);
#else
    asm volatile("s_wait_asynccnt 0" ::: "memory");
#endif
#endif
}

// ---------------------------------------------------------------------------
// f32 -> bf16 bulk conversion (8 elements/thread, b128 in / b128 out)
// ---------------------------------------------------------------------------
__global__ __launch_bounds__(256)
void cvt_bf16_kernel(const float* __restrict__ X, unsigned short* __restrict__ Y,
                     long long n)
{
    const long long i = ((long long)blockIdx.x * 256 + threadIdx.x) * 8;
    if (i >= n) return;
    const f32x4 a = *(const f32x4*)(X + i);
    const f32x4 b = *(const f32x4*)(X + i + 4);
    u32x4 o;
    o.x = rne2(a.x, a.y); o.y = rne2(a.z, a.w);
    o.z = rne2(b.x, b.y); o.w = rne2(b.z, b.w);
    *(u32x4*)(Y + i) = o;
}

// ---------------------------------------------------------------------------
// GEMM: C[M,N] = act(A[M,K] @ W[N,K]^T + bias[N]); A,W bf16; C f32 or bf16.
// Block tile 128x128, K-step 32, 8 waves, wave tile 64x32 (4x2 WMMA tiles).
// Double-buffered async LDS staging: copy tile t+1 while computing tile t.
// ---------------------------------------------------------------------------
#define LDSS 48   // ushorts per LDS row (32 data + 16 pad) = 96B, 16B aligned

template<int OUTBF, int RELU>
__global__ __launch_bounds__(256)
void gemm_bf16_kernel(const unsigned short* __restrict__ A,
                      const unsigned short* __restrict__ W,
                      const float* __restrict__ bias, void* __restrict__ Cout,
                      int M, int N, int K)
{
    __shared__ unsigned short lA[2][128 * LDSS];
    __shared__ unsigned short lB[2][128 * LDSS];

    const int tid  = threadIdx.x;
    const int wave = tid >> 5;
    const int lane = tid & 31;
    const int l16  = lane & 15;
    const int hi   = lane >> 4;

    const int bm = blockIdx.x * 128;
    const int bn = blockIdx.y * 128;
    const int wm = (wave >> 2) * 64;
    const int wn = (wave & 3) * 32;

    v8f acc[4][2];
#pragma unroll
    for (int mi = 0; mi < 4; ++mi)
#pragma unroll
        for (int nj = 0; nj < 2; ++nj) acc[mi][nj] = {};

    const int cpRow = tid >> 1;         // 0..127
    const int cpCol = (tid & 1) * 16;   // ushort col base (two 16B chunks)

    auto stage = [&](int k0, int buf) {
        const unsigned short* ga = A + (size_t)(bm + cpRow) * K + k0 + cpCol;
        const unsigned short* gw = W + (size_t)(bn + cpRow) * K + k0 + cpCol;
        unsigned short* la = lA[buf] + cpRow * LDSS + cpCol;
        unsigned short* lb = lB[buf] + cpRow * LDSS + cpCol;
        cp16(la,     ga);
        cp16(la + 8, ga + 8);
        cp16(lb,     gw);
        cp16(lb + 8, gw + 8);
    };

    stage(0, 0);
    const int nk = K >> 5;
    for (int kt = 0; kt < nk; ++kt) {
        async_join();          // tile kt landed in lA/lB[cur]
        __syncthreads();
        const int cur = kt & 1;
        if (kt + 1 < nk) stage((kt + 1) << 5, cur ^ 1);   // fill other buffer
        if (kt + 2 < nk) {     // warm L2 two tiles ahead
            __builtin_prefetch(A + (size_t)(bm + cpRow) * K + ((kt + 2) << 5) + cpCol, 0, 1);
            __builtin_prefetch(W + (size_t)(bn + cpRow) * K + ((kt + 2) << 5) + cpCol, 0, 1);
        }

        v16bf aop[4];
#pragma unroll
        for (int mi = 0; mi < 4; ++mi) {
            const unsigned short* base = lA[cur] + (wm + mi * 16 + l16) * LDSS;
            V16U t;
            t.q[0] = *(const u32x4*)(base + (hi ? 8  : 0));   // K 0..7  / 8..15
            t.q[1] = *(const u32x4*)(base + (hi ? 24 : 16));  // K16..23 / 24..31
            aop[mi] = t.v;
        }
        v16bf bop[2];
#pragma unroll
        for (int nj = 0; nj < 2; ++nj) {
            const unsigned short* base = lB[cur] + (wn + nj * 16 + l16) * LDSS;
            V16U t;
            t.q[0] = *(const u32x4*)(base + (hi ? 8  : 0));
            t.q[1] = *(const u32x4*)(base + (hi ? 24 : 16));
            bop[nj] = t.v;
        }
#pragma unroll
        for (int mi = 0; mi < 4; ++mi)
#pragma unroll
            for (int nj = 0; nj < 2; ++nj)
                acc[mi][nj] = wmma_bf16(aop[mi], bop[nj], acc[mi][nj]);
    }

#pragma unroll
    for (int nj = 0; nj < 2; ++nj) {
        const int n  = bn + wn + nj * 16 + l16;
        const float bv = bias[n];
#pragma unroll
        for (int mi = 0; mi < 4; ++mi) {
#pragma unroll
            for (int i = 0; i < 8; ++i) {
                const int row = bm + wm + mi * 16 + (hi ? i + 8 : i);
                float v = acc[mi][nj][i] + bv;
                if (RELU) v = v > 0.f ? v : 0.f;
                if (OUTBF) ((unsigned short*)Cout)[(size_t)row * N + n] = f2bf(v);
                else       ((float*)Cout)[(size_t)row * N + n] = v;
            }
        }
    }
}

// ---------------------------------------------------------------------------
// Flash attention over bf16 Q/K/V, bf16 output. grid (B*H, T/64), 4 waves.
// Double-buffered K/V tiles: stage tile t+1 while computing tile t.
// ---------------------------------------------------------------------------
__global__ __launch_bounds__(128)
void flash_attn_kernel(const unsigned short* __restrict__ Qb,
                       const unsigned short* __restrict__ Kb,
                       const unsigned short* __restrict__ Vb,
                       const int* __restrict__ mask,
                       unsigned short* __restrict__ Ob, int T, int H)
{
    __shared__ unsigned short sK [2][32 * 80];   // [key][d], stride 80 (160B)
    __shared__ unsigned short sVt[2][64 * 48];   // [d][key], stride 48 (96B)
    __shared__ unsigned short sP [4][16 * 32];   // per-wave P scratch

    const int tid  = threadIdx.x;
    const int wave = tid >> 5;
    const int lane = tid & 31;
    const int l16  = lane & 15;
    const int hi   = lane >> 4;

    const int bh    = blockIdx.x;
    const int b     = bh / H;
    const int h     = bh % H;
    const int qtile = blockIdx.y * 64 + wave * 16;
    const int D     = H * 64;
    const size_t headOff = (size_t)b * T * D + (size_t)h * 64;

    // Q tile -> registers, A layout (bf16, direct b128 loads)
    v16bf qa[2];
    {
        const unsigned short* qrow = Qb + headOff + (size_t)(qtile + l16) * D;
#pragma unroll
        for (int c = 0; c < 2; ++c) {
            V16U t;
            t.q[0] = *(const u32x4*)(qrow + 32 * c + (hi ? 8 : 0));
            t.q[1] = *(const u32x4*)(qrow + 32 * c + 16 + (hi ? 8 : 0));
            qa[c] = t.v;
        }
    }

    v8f o[4];
#pragma unroll
    for (int jd = 0; jd < 4; ++jd) o[jd] = {};
    float rmax[8], rsum[8];
#pragma unroll
    for (int i = 0; i < 8; ++i) { rmax[i] = -1e30f; rsum[i] = 0.f; }

    const float scale = 0.125f;  // 1/sqrt(64)
    const int cpR = tid >> 2;          // key row 0..31
    const int cpC = (tid & 3) * 16;    // ushort col base

    auto stageKV = [&](int kb, int buf) {
        const unsigned short* kg = Kb + headOff + (size_t)(kb + cpR) * D + cpC;
        const unsigned short* vg = Vb + headOff + (size_t)(kb + cpR) * D + cpC;
        unsigned short* lk = sK[buf] + cpR * 80 + cpC;
        cp16(lk,     kg);          // K: pure async copy
        cp16(lk + 8, kg + 8);
        U8x16 u0, u1;              // V: transpose through VGPRs
        u0.q = *(const u32x4*)(vg);
        u1.q = *(const u32x4*)(vg + 8);
#pragma unroll
        for (int j = 0; j < 8; ++j) {
            sVt[buf][(cpC + j) * 48 + cpR]     = u0.s[j];
            sVt[buf][(cpC + 8 + j) * 48 + cpR] = u1.s[j];
        }
    };

    stageKV(0, 0);
    const int nkt = T / 32;
    for (int kt = 0; kt < nkt; ++kt) {
        const int kb = kt * 32;
        async_join();          // tile kt landed
        __syncthreads();
        const int cur = kt & 1;
        if (kt + 1 < nkt) stageKV(kb + 32, cur ^ 1);

        // batch-issue mask loads so one wait covers all 16
        int mv[16];
#pragma unroll
        for (int i = 0; i < 8; ++i) {
            const size_t mrow = (size_t)(qtile + (hi ? i + 8 : i)) * T + kb + l16;
            mv[i]     = mask[mrow];
            mv[8 + i] = mask[mrow + 16];
        }

        // S = Q K^T : two 16x16 tiles
        v8f s0 = {}, s1 = {};
#pragma unroll
        for (int c = 0; c < 2; ++c) {
            const unsigned short* b0 = sK[cur] + l16 * 80        + 32 * c + (hi ? 8 : 0);
            const unsigned short* b1 = sK[cur] + (16 + l16) * 80 + 32 * c + (hi ? 8 : 0);
            V16U t0, t1;
            t0.q[0] = *(const u32x4*)(b0); t0.q[1] = *(const u32x4*)(b0 + 16);
            t1.q[0] = *(const u32x4*)(b1); t1.q[1] = *(const u32x4*)(b1 + 16);
            s0 = wmma_bf16(qa[c], t0.v, s0);
            s1 = wmma_bf16(qa[c], t1.v, s1);
        }

        // online softmax; row reductions via DPP ROW_XMASK (no LDS traffic)
        float p0[8], p1[8];
#pragma unroll
        for (int i = 0; i < 8; ++i) {
            float a0 = s0[i] * scale;
            float a1 = s1[i] * scale;
            if (mv[i]     == 0) a0 = -10000.f;
            if (mv[8 + i] == 0) a1 = -10000.f;
            const float mx   = row16_max(fmaxf(a0, a1));
            const float mnew = fmaxf(rmax[i], mx);
            const float corr = __expf(rmax[i] - mnew);
            const float e0 = __expf(a0 - mnew);
            const float e1 = __expf(a1 - mnew);
            const float ps = row16_sum(e0 + e1);
            rsum[i] = rsum[i] * corr + ps;
            rmax[i] = mnew;
#pragma unroll
            for (int jd = 0; jd < 4; ++jd) o[jd][i] *= corr;
            p0[i] = e0; p1[i] = e1;
        }

        // P (C layout) -> per-wave LDS -> A layout (K = 32 keys)
        unsigned short* pw = sP[wave];
#pragma unroll
        for (int i = 0; i < 8; ++i) {
            const int rl = hi ? i + 8 : i;
            pw[rl * 32 + l16]      = f2bf(p0[i]);
            pw[rl * 32 + 16 + l16] = f2bf(p1[i]);
        }
        V16U tp;
        const unsigned short* pb = pw + l16 * 32 + (hi ? 8 : 0);
        tp.q[0] = *(const u32x4*)(pb);
        tp.q[1] = *(const u32x4*)(pb + 16);
        const v16bf pa = tp.v;

        // O += P V
#pragma unroll
        for (int jd = 0; jd < 4; ++jd) {
            const unsigned short* vb = sVt[cur] + (jd * 16 + l16) * 48 + (hi ? 8 : 0);
            V16U tv;
            tv.q[0] = *(const u32x4*)(vb);
            tv.q[1] = *(const u32x4*)(vb + 16);
            o[jd] = wmma_bf16(pa, tv.v, o[jd]);
        }
    }

    // normalize and store bf16 ctx[b, q, h, d]
#pragma unroll
    for (int jd = 0; jd < 4; ++jd) {
        const int d = jd * 16 + l16;
#pragma unroll
        for (int i = 0; i < 8; ++i) {
            const int row = qtile + (hi ? i + 8 : i);
            Ob[headOff + (size_t)row * D + d] = f2bf(o[jd][i] / rsum[i]);
        }
    }
}

// ---------------------------------------------------------------------------
// Fused residual add + LayerNorm; optional dual f32+bf16 output.
// ---------------------------------------------------------------------------
__global__ __launch_bounds__(256)
void add_ln_kernel(const float* __restrict__ X, const float* __restrict__ R,
                   const float* __restrict__ g, const float* __restrict__ be,
                   float* __restrict__ Yf, unsigned short* __restrict__ Ybf, int D)
{
    __shared__ float red[16];
    const int row = blockIdx.x;
    const int tid = threadIdx.x;
    const float* x = X + (size_t)row * D;
    const float* r = R + (size_t)row * D;

    float v[4], s = 0.f, ss = 0.f;
#pragma unroll
    for (int j = 0; j < 4; ++j) {
        const int idx = tid + j * 256;
        const float t = x[idx] + r[idx];
        v[j] = t; s += t; ss += t * t;
    }
#pragma unroll
    for (int d = 16; d >= 1; d >>= 1) { s += __shfl_xor(s, d, 32); ss += __shfl_xor(ss, d, 32); }
    const int wave = tid >> 5, lane = tid & 31;
    if (lane == 0) { red[wave] = s; red[8 + wave] = ss; }
    __syncthreads();
    if (wave == 0) {
        float a  = lane < 8 ? red[lane] : 0.f;
        float b2 = lane < 8 ? red[8 + lane] : 0.f;
#pragma unroll
        for (int d = 4; d >= 1; d >>= 1) { a += __shfl_xor(a, d, 32); b2 += __shfl_xor(b2, d, 32); }
        if (lane == 0) { red[0] = a; red[1] = b2; }
    }
    __syncthreads();
    const float inv  = 1.0f / (float)D;
    const float mean = red[0] * inv;
    const float var  = red[1] * inv - mean * mean;
    const float rstd = rsqrtf(var + 1e-12f);
#pragma unroll
    for (int j = 0; j < 4; ++j) {
        const int idx = tid + j * 256;
        const float y = g[idx] * (v[j] - mean) * rstd + be[idx];
        Yf[(size_t)row * D + idx] = y;
        if (Ybf) Ybf[(size_t)row * D + idx] = f2bf(y);
    }
}

// ---------------------------------------------------------------------------
// Launcher
// ---------------------------------------------------------------------------
extern "C" void kernel_launch(void* const* d_in, const int* in_sizes, int n_in,
                              void* d_out, int out_size, void* d_ws, size_t ws_size,
                              hipStream_t stream) {
    const float* dec    = (const float*)d_in[0];
    const float* enc    = (const float*)d_in[1];
    const int*   t_mask = (const int*)d_in[2];
    const int*   s_mask = (const int*)d_in[3];
    const float* wsrc[10] = {
        (const float*)d_in[4],  (const float*)d_in[6],  (const float*)d_in[8],
        (const float*)d_in[10], (const float*)d_in[12], (const float*)d_in[14],
        (const float*)d_in[16], (const float*)d_in[18], (const float*)d_in[20],
        (const float*)d_in[22] };
    const float* sa_bq = (const float*)d_in[5],  *sa_bk = (const float*)d_in[7];
    const float* sa_bv = (const float*)d_in[9],  *sa_bo = (const float*)d_in[11];
    const float* ca_bq = (const float*)d_in[13], *ca_bk = (const float*)d_in[15];
    const float* ca_bv = (const float*)d_in[17], *ca_bo = (const float*)d_in[19];
    const float* b1 = (const float*)d_in[21], *b2 = (const float*)d_in[23];
    const float* g1 = (const float*)d_in[24], *be1 = (const float*)d_in[25];
    const float* g2 = (const float*)d_in[26], *be2 = (const float*)d_in[27];
    const float* g3 = (const float*)d_in[28], *be3 = (const float*)d_in[29];

    const int B = 2, T = 2048, D = 1024, H = 16, F = 4096;
    const size_t MT = (size_t)B * T;   // 4096 rows

    // ---- workspace carve (256B aligned) ----
    char* wp = (char*)d_ws;
    auto carve = [&](size_t bytes) {
        char* p = wp; wp += (bytes + 255) & ~(size_t)255; return p;
    };
    unsigned short* decbf = (unsigned short*)carve(MT * D * 2);
    unsigned short* encbf = (unsigned short*)carve(MT * D * 2);
    unsigned short* wbf[10];
    const size_t welems[10] = { (size_t)D*D, (size_t)D*D, (size_t)D*D, (size_t)D*D,
                                (size_t)D*D, (size_t)D*D, (size_t)D*D, (size_t)D*D,
                                (size_t)F*D, (size_t)D*F };
    for (int i = 0; i < 10; ++i) wbf[i] = (unsigned short*)carve(welems[i] * 2);
    unsigned short* Qb    = (unsigned short*)carve(MT * D * 2);
    unsigned short* Kb    = (unsigned short*)carve(MT * D * 2);
    unsigned short* Vb    = (unsigned short*)carve(MT * D * 2);
    unsigned short* CTXbf = (unsigned short*)carve(MT * D * 2);
    unsigned short* X1bf  = (unsigned short*)carve(MT * D * 2);
    unsigned short* X2bf  = (unsigned short*)carve(MT * D * 2);
    unsigned short* Hbf   = (unsigned short*)carve(MT * F * 2);
    float* TMP = (float*)carve(MT * D * 4);
    float* X1  = (float*)carve(MT * D * 4);
    float* X2  = (float*)carve(MT * D * 4);

    const dim3 blk256(256), blk128(128);
    const dim3 gDD((unsigned)(MT / 128), D / 128);   // (32, 8)
    const dim3 gDF((unsigned)(MT / 128), F / 128);   // (32, 32)
    const dim3 gAT((unsigned)(B * H), T / 64);       // (32, 32)

    auto cvt = [&](const float* src, unsigned short* dst, size_t n) {
        cvt_bf16_kernel<<<dim3((unsigned)((n / 8 + 255) / 256)), blk256, 0, stream>>>(
            src, dst, (long long)n);
    };

    // ---- one-time bf16 conversions (weights + encoder/decoder inputs) ----
    cvt(dec, decbf, MT * D);
    cvt(enc, encbf, MT * D);
    for (int i = 0; i < 10; ++i) cvt(wsrc[i], wbf[i], welems[i]);

    const unsigned short *sa_wq = wbf[0], *sa_wk = wbf[1], *sa_wv = wbf[2], *sa_wo = wbf[3];
    const unsigned short *ca_wq = wbf[4], *ca_wk = wbf[5], *ca_wv = wbf[6], *ca_wo = wbf[7];
    const unsigned short *w1bf = wbf[8], *w2bf = wbf[9];

    // ---- self-attention block ----
    gemm_bf16_kernel<1,0><<<gDD, blk256, 0, stream>>>(decbf, sa_wq, sa_bq, Qb, (int)MT, D, D);
    gemm_bf16_kernel<1,0><<<gDD, blk256, 0, stream>>>(decbf, sa_wk, sa_bk, Kb, (int)MT, D, D);
    gemm_bf16_kernel<1,0><<<gDD, blk256, 0, stream>>>(decbf, sa_wv, sa_bv, Vb, (int)MT, D, D);
    flash_attn_kernel<<<gAT, blk128, 0, stream>>>(Qb, Kb, Vb, t_mask, CTXbf, T, H);
    gemm_bf16_kernel<0,0><<<gDD, blk256, 0, stream>>>(CTXbf, sa_wo, sa_bo, TMP, (int)MT, D, D);
    add_ln_kernel<<<dim3((unsigned)MT), blk256, 0, stream>>>(TMP, dec, g1, be1, X1, X1bf, D);

    // ---- cross-attention block ----
    gemm_bf16_kernel<1,0><<<gDD, blk256, 0, stream>>>(X1bf,  ca_wq, ca_bq, Qb, (int)MT, D, D);
    gemm_bf16_kernel<1,0><<<gDD, blk256, 0, stream>>>(encbf, ca_wk, ca_bk, Kb, (int)MT, D, D);
    gemm_bf16_kernel<1,0><<<gDD, blk256, 0, stream>>>(encbf, ca_wv, ca_bv, Vb, (int)MT, D, D);
    flash_attn_kernel<<<gAT, blk128, 0, stream>>>(Qb, Kb, Vb, s_mask, CTXbf, T, H);
    gemm_bf16_kernel<0,0><<<gDD, blk256, 0, stream>>>(CTXbf, ca_wo, ca_bo, TMP, (int)MT, D, D);
    add_ln_kernel<<<dim3((unsigned)MT), blk256, 0, stream>>>(TMP, X1, g2, be2, X2, X2bf, D);

    // ---- FFN block ----
    gemm_bf16_kernel<1,1><<<gDF, blk256, 0, stream>>>(X2bf, w1bf, b1, Hbf, (int)MT, F, D);
    gemm_bf16_kernel<0,0><<<gDD, blk256, 0, stream>>>(Hbf,  w2bf, b2, TMP, (int)MT, D, F);
    add_ln_kernel<<<dim3((unsigned)MT), blk256, 0, stream>>>(TMP, X2, g3, be3,
                                                             (float*)d_out, nullptr, D);
}
